// SAGE_Re_58308476011190
// MI455X (gfx1250) — compile-verified, matrix operand order checked by reading, not run on hardware
//
#include <hip/hip_runtime.h>
#include <hip/hip_bf16.h>

typedef __attribute__((ext_vector_type(16))) __bf16 v16bf;
typedef __attribute__((ext_vector_type(8)))  __bf16 v8bf;
typedef __attribute__((ext_vector_type(4)))  __bf16 v4bf;
typedef __attribute__((ext_vector_type(8)))  float  v8f;
typedef __attribute__((ext_vector_type(4)))  unsigned int v4u;
typedef __attribute__((ext_vector_type(4)))  int    v4i;
typedef __attribute__((ext_vector_type(8)))  int    v8i;

#ifndef __has_builtin
#define __has_builtin(x) 0
#endif
#if __has_builtin(__builtin_amdgcn_tensor_load_to_lds)
#define HAVE_TDM 1
#else
#define HAVE_TDM 0
#endif

// ---------------------------------------------------------------------------
// Utility kernels
// ---------------------------------------------------------------------------

__global__ void k_zero(float* __restrict__ p, long n) {
    long j = ((long)blockIdx.x * blockDim.x + threadIdx.x) * 4;
    if (j + 3 < n) {
        *(float4*)(p + j) = make_float4(0.f, 0.f, 0.f, 0.f);
    } else {
        for (long k = j; k < n; ++k) p[k] = 0.f;
    }
}

__global__ void k_degree(const int* __restrict__ ei, float* __restrict__ deg, int E) {
    int e = blockIdx.x * blockDim.x + threadIdx.x;
    if (e < E) atomicAdd(&deg[ei[e]], 1.0f);
}

__global__ void k_norms(const float* __restrict__ deg, float* __restrict__ dinv,
                        float* __restrict__ cinv, int n) {
    int i = blockIdx.x * blockDim.x + threadIdx.x;
    if (i >= n) return;
    float d = deg[i];
    dinv[i] = (d > 0.f) ? rsqrtf(d) : 0.f;
    cinv[i] = 1.f / fmaxf(d, 1.f);
}

// scatter-add: out[row] += h[col] * (colScale ? colScale[col] : 1)
template <int C>
__global__ void k_scatter(const int* __restrict__ ei, const float* __restrict__ h,
                          const float* __restrict__ colScale, float* __restrict__ out, int E) {
    constexpr int G = C / 4;
    long t = (long)blockIdx.x * blockDim.x + threadIdx.x;
    int e  = (int)(t / G);
    int c4 = (int)(t % G) * 4;
    if (e >= E) return;
    int row = ei[e];
    int col = ei[(long)E + e];
    float s = colScale ? colScale[col] : 1.0f;
    float4 v = *(const float4*)(h + (size_t)col * C + c4);
    float* o = out + (size_t)row * C + c4;
    atomicAdd(o + 0, v.x * s);
    atomicAdd(o + 1, v.y * s);
    atomicAdd(o + 2, v.z * s);
    atomicAdd(o + 3, v.w * s);
}

// out_bf16[i,c] = bf16(a[i,c] * rowScale[i])
template <int C>
__global__ void k_scale_cvt(const float* __restrict__ a, const float* __restrict__ rowScale,
                            __bf16* __restrict__ out, int n) {
    constexpr int G = C / 4;
    long t = (long)blockIdx.x * blockDim.x + threadIdx.x;
    int i  = (int)(t / G);
    int c4 = (int)(t % G) * 4;
    if (i >= n) return;
    float s  = rowScale[i];
    float4 v = *(const float4*)(a + (size_t)i * C + c4);
    v4bf o   = { (__bf16)(v.x * s), (__bf16)(v.y * s), (__bf16)(v.z * s), (__bf16)(v.w * s) };
    *(v4bf*)(out + (size_t)i * C + c4) = o;
}

// plain fp32 -> bf16 convert
__global__ void k_cvt(const float* __restrict__ in, __bf16* __restrict__ out, long n) {
    long j = ((long)blockIdx.x * blockDim.x + threadIdx.x) * 4;
    if (j + 3 >= n) {
        for (long k = j; k < n; ++k) out[k] = (__bf16)in[k];
        return;
    }
    float4 v = *(const float4*)(in + j);
    v4bf o = { (__bf16)v.x, (__bf16)v.y, (__bf16)v.z, (__bf16)v.w };
    *(v4bf*)(out + j) = o;
}

// Wt[cout*K + cin] = bf16(W[cin*COUT + cout])   (weights tiny; once per call)
__global__ void k_wtrans(const float* __restrict__ W, __bf16* __restrict__ Wt, int K, int COUT) {
    int t = blockIdx.x * blockDim.x + threadIdx.x;
    if (t >= K * COUT) return;
    int cin = t / COUT, cout = t % COUT;
    Wt[(size_t)cout * K + cin] = (__bf16)W[t];
}

// ---------------------------------------------------------------------------
// Fused WMMA GEMM with TDM-staged weights:
//   out = epilogue( A @ W^T  [+ H @ R^T]  + bias )
// Block = 256 threads (8 waves). blockIdx.x selects an 8-row-tile group,
// blockIdx.y selects the column group (CT 16-wide tiles). The block's weight
// slice [CT*16 x K] is DMA'd into LDS by the Tensor Data Mover with LDS row
// padding of 8 bf16 (-> row stride K+8 elems; ds_load_b128 bank stride = 4
// dwords, conflict-free). A row-slices live in registers; the inner loop is
// ds_load_b128 + v_wmma_f32_16x16x32_bf16 with f32 accumulation.
// ---------------------------------------------------------------------------
template <int K, int COUT, int CT, bool SECOND, bool RELU, bool REZERO>
__global__ void k_wmma_gemm(const __bf16* __restrict__ A, const __bf16* __restrict__ Wt,
                            const __bf16* __restrict__ H, const __bf16* __restrict__ Rt,
                            const float* __restrict__ bias,
                            const float* __restrict__ alpha, int alpha_idx,
                            const float* __restrict__ resid,
                            float* __restrict__ out, int nrows) {
    constexpr int ROWS = CT * 16;   // weight rows (= output cols) staged per block
    constexpr int KP   = K + 8;     // padded LDS row stride (elements)
    __shared__ __bf16 sm[ROWS * KP];

    const int wv   = threadIdx.x >> 5;
    const int lane = threadIdx.x & 31;
    const int tileRow = blockIdx.x * 8 + wv;
    const int cg      = blockIdx.y;
    const bool active = (tileRow * 16 < nrows);
    const int lr = lane & 15;   // A row within tile / B col within tile
    const int hi = lane >> 4;   // K-half select (kb = hi*8)

    // ---- stage a [ROWS x K] bf16 weight slice into LDS ----
    auto stage = [&](const __bf16* __restrict__ src) {
#if HAVE_TDM
        if ((threadIdx.x >> 5) == 0) {
            unsigned lds_off = (unsigned)(size_t)(void*)sm;
            unsigned long long ga = (unsigned long long)(size_t)src;
            v4u g0 = { 1u,                                     // count=1, user mode
                       lds_off,                                // lds_addr
                       (unsigned)(ga & 0xFFFFFFFFu),           // global_addr[31:0]
                       (unsigned)((ga >> 32) & 0x01FFFFFFu)    // global_addr[56:32]
                           | (2u << 30) };                     // type=2 (image)
            // pad every K/2 dwords (one row) by 4 dwords (8 bf16)
            constexpr unsigned PI = (K == 256) ? 6u : 5u;      // 2^(PI+1) dwords
            v8i g1 = { (int)((1u << 16)                        // data_size = 2B
                           | (1u << 20)                        // pad_enable
                           | (PI << 22)                        // pad_interval
                           | (3u << 25)),                      // pad_amount = 4 dwords
                       (int)((unsigned)K << 16),               // tensor_dim0[15:0] @ bits63:48
                       (int)((unsigned)ROWS << 16),            // tensor_dim1[15:0] @ bits95:80
                       (int)((unsigned)K << 16),               // tile_dim0 @ bits127:112
                       (int)ROWS,                              // tile_dim1 @ bits143:128
                       (int)K,                                 // tensor_dim0_stride low32
                       0, 0 };
            v4i gz = { 0, 0, 0, 0 };
#if __clang_major__ >= 23
            v8i gz8 = { 0, 0, 0, 0, 0, 0, 0, 0 };
            __builtin_amdgcn_tensor_load_to_lds(g0, g1, gz, gz, gz8, 0);
#else
            __builtin_amdgcn_tensor_load_to_lds(g0, g1, gz, gz, 0);
#endif
            __builtin_amdgcn_s_wait_tensorcnt(0);
        }
#else
        constexpr int CPR = K / 8;  // 16B chunks per row
        for (int idx = threadIdx.x; idx < ROWS * CPR; idx += 256) {
            int r = idx / CPR, c = (idx % CPR) * 8;
            *(v8bf*)&sm[r * KP + c] = *(const v8bf*)(src + (size_t)r * K + c);
        }
#endif
        __syncthreads();
    };

    v8f acc[CT] = {};
    const size_t rowOff = (size_t)(tileRow * 16 + lr) * K + hi * 8;

    // One accumulation pass: preload the A row-slice into registers, then
    // run the pure ds_load + wmma loop over K.
    auto pass = [&](const __bf16* __restrict__ ap) {
        v8bf areg[K / 16];
#pragma unroll
        for (int s = 0; s < K / 32; ++s) {
            areg[2 * s]     = *(const v8bf*)(ap + s * 32);
            areg[2 * s + 1] = *(const v8bf*)(ap + s * 32 + 16);
        }
#pragma unroll
        for (int s = 0; s < K / 32; ++s) {
            v16bf a = __builtin_shufflevector(areg[2 * s], areg[2 * s + 1],
                                              0, 1, 2, 3, 4, 5, 6, 7,
                                              8, 9, 10, 11, 12, 13, 14, 15);
            const int k0 = s * 32;
#pragma unroll
            for (int t = 0; t < CT; ++t) {
                const __bf16* bp = &sm[(t * 16 + lr) * KP + hi * 8 + k0];
                v8bf blo = *(const v8bf*)(bp);
                v8bf bhi = *(const v8bf*)(bp + 16);
                v16bf b = __builtin_shufflevector(blo, bhi, 0, 1, 2, 3, 4, 5, 6, 7,
                                                  8, 9, 10, 11, 12, 13, 14, 15);
                acc[t] = __builtin_amdgcn_wmma_f32_16x16x32_bf16(
                    false, a, false, b, (short)0, acc[t], false, false);
            }
        }
    };

    // pass 1: A @ W^T
    stage(Wt + (size_t)cg * ROWS * K);
    if (active) pass(A + rowOff);

    // pass 2 (SAGE root term): H @ R^T into the same accumulators
    if constexpr (SECOND) {
        __syncthreads();                       // everyone done reading W slice
        stage(Rt + (size_t)cg * ROWS * K);
        if (active) pass(H + rowOff);
    }

    if (!active) return;

    float av = 0.f;
    if constexpr (REZERO) av = alpha[alpha_idx];

    // C/D layout: lanes 0-15 hold M = v, lanes 16-31 hold M = v+8; N = lane&15
#pragma unroll
    for (int t = 0; t < CT; ++t) {
        int c = (cg * CT + t) * 16 + lr;
        float bv = bias[c];
#pragma unroll
        for (int v = 0; v < 8; ++v) {
            int r = tileRow * 16 + v + hi * 8;
            float val = acc[t][v] + bv;
            if constexpr (RELU) val = fmaxf(val, 0.f);
            if constexpr (REZERO) val = resid[(size_t)r * COUT + c] + av * val;
            out[(size_t)r * COUT + c] = val;
        }
    }
}

// ---------------------------------------------------------------------------
// Host-side orchestration
// ---------------------------------------------------------------------------

static inline long cdivl(long a, long b) { return (a + b - 1) / b; }

extern "C" void kernel_launch(void* const* d_in, const int* in_sizes, int n_in,
                              void* d_out, int out_size, void* d_ws, size_t ws_size,
                              hipStream_t stream) {
    const float* x     = (const float*)d_in[0];
    const int*   ei    = (const int*)d_in[1];
    const float* alpha = (const float*)d_in[2];
    const float* W0 = (const float*)d_in[3];
    const float* b0 = (const float*)d_in[4];
    const float* W1 = (const float*)d_in[5];
    const float* R1 = (const float*)d_in[6];
    const float* b1 = (const float*)d_in[7];
    const float* W2 = (const float*)d_in[8];
    const float* R2 = (const float*)d_in[9];
    const float* b2 = (const float*)d_in[10];
    const float* W3 = (const float*)d_in[11];
    const float* b3 = (const float*)d_in[12];
    const float* W4 = (const float*)d_in[13];
    const float* R4 = (const float*)d_in[14];
    const float* b4 = (const float*)d_in[15];

    const int N = in_sizes[0] / 128;   // 100000 (divisible by 16)
    const int E = in_sizes[1] / 2;     // 800000

    // -- workspace partition (256B aligned) --
    char* w = (char*)d_ws;
    auto alloc = [&](size_t bytes) -> void* {
        void* p = (void*)w;
        w += (bytes + 255) & ~(size_t)255;
        return p;
    };
    float* deg  = (float*)alloc((size_t)N * 4);
    float* dinv = (float*)alloc((size_t)N * 4);
    float* cinv = (float*)alloc((size_t)N * 4);
    float* AGG  = (float*)alloc((size_t)N * 256 * 4);
    float* Ha   = (float*)alloc((size_t)N * 256 * 4);   // h0 (stride128) then h2
    float* Hb   = (float*)alloc((size_t)N * 256 * 4);   // h1 then h3
    __bf16* Abf = (__bf16*)alloc((size_t)N * 256 * 2);
    __bf16* Hbf = (__bf16*)alloc((size_t)N * 256 * 2);
    __bf16* W0t = (__bf16*)alloc(128 * 128 * 2);
    __bf16* W1t = (__bf16*)alloc(128 * 256 * 2);
    __bf16* R1t = (__bf16*)alloc(128 * 256 * 2);
    __bf16* W2t = (__bf16*)alloc(256 * 256 * 2);
    __bf16* R2t = (__bf16*)alloc(256 * 256 * 2);
    __bf16* W3t = (__bf16*)alloc(256 * 256 * 2);
    __bf16* W4t = (__bf16*)alloc(256 * 112 * 2);
    __bf16* R4t = (__bf16*)alloc(256 * 112 * 2);

    const int B = 256;
    auto zero = [&](float* p, long n) {
        k_zero<<<(int)cdivl(n, 4L * B), B, 0, stream>>>(p, n);
    };
    auto wtrans = [&](const float* Wp, __bf16* Wtp, int K, int COUT) {
        k_wtrans<<<(int)cdivl((long)K * COUT, B), B, 0, stream>>>(Wp, Wtp, K, COUT);
    };
    const int rowGroups = (int)cdivl((N + 15) / 16, 8);  // 8 row-tiles per block

    // -- graph prep: degree, normalizers, bf16 weights --
    zero(deg, N);
    k_degree<<<(int)cdivl(E, B), B, 0, stream>>>(ei, deg, E);
    k_norms<<<(int)cdivl(N, B), B, 0, stream>>>(deg, dinv, cinv, N);
    wtrans(W0, W0t, 128, 128);
    wtrans(W1, W1t, 128, 256);
    wtrans(R1, R1t, 128, 256);
    wtrans(W2, W2t, 256, 256);
    wtrans(R2, R2t, 256, 256);
    wtrans(W3, W3t, 256, 256);
    wtrans(W4, W4t, 256, 112);
    wtrans(R4, R4t, 256, 112);

    // ---- layer 0: GCN, ReZero.  h0 = x + alpha0 * (gcn_agg(x) @ W0 + b0) ----
    zero(AGG, (long)N * 128);
    k_scatter<128><<<(int)cdivl((long)E * 32, B), B, 0, stream>>>(ei, x, dinv, AGG, E);
    k_scale_cvt<128><<<(int)cdivl((long)N * 32, B), B, 0, stream>>>(AGG, dinv, Abf, N);
    k_wmma_gemm<128, 128, 4, false, false, true><<<dim3(rowGroups, 2), B, 0, stream>>>(
        Abf, W0t, nullptr, nullptr, b0, alpha, 0, x, Ha, N);

    // ---- layer 1: SAGE+ReLU. h1 = relu(mean(h0)@W1 + h0@R1 + b1) ----
    zero(AGG, (long)N * 128);
    k_scatter<128><<<(int)cdivl((long)E * 32, B), B, 0, stream>>>(ei, Ha, nullptr, AGG, E);
    k_scale_cvt<128><<<(int)cdivl((long)N * 32, B), B, 0, stream>>>(AGG, cinv, Abf, N);
    k_cvt<<<(int)cdivl((long)N * 128, 4L * B), B, 0, stream>>>(Ha, Hbf, (long)N * 128);
    k_wmma_gemm<128, 256, 4, true, true, false><<<dim3(rowGroups, 4), B, 0, stream>>>(
        Abf, W1t, Hbf, R1t, b1, nullptr, 0, nullptr, Hb, N);

    // ---- layer 2: SAGE+ReLU. h2 = relu(mean(h1)@W2 + h1@R2 + b2) ----
    zero(AGG, (long)N * 256);
    k_scatter<256><<<(int)cdivl((long)E * 64, B), B, 0, stream>>>(ei, Hb, nullptr, AGG, E);
    k_scale_cvt<256><<<(int)cdivl((long)N * 64, B), B, 0, stream>>>(AGG, cinv, Abf, N);
    k_cvt<<<(int)cdivl((long)N * 256, 4L * B), B, 0, stream>>>(Hb, Hbf, (long)N * 256);
    k_wmma_gemm<256, 256, 4, true, true, false><<<dim3(rowGroups, 4), B, 0, stream>>>(
        Abf, W2t, Hbf, R2t, b2, nullptr, 0, nullptr, Ha, N);

    // ---- layer 3: GCN, ReZero. h3 = h2 + alpha3 * (gcn_agg(h2) @ W3 + b3) ----
    zero(AGG, (long)N * 256);
    k_scatter<256><<<(int)cdivl((long)E * 64, B), B, 0, stream>>>(ei, Ha, dinv, AGG, E);
    k_scale_cvt<256><<<(int)cdivl((long)N * 64, B), B, 0, stream>>>(AGG, dinv, Abf, N);
    k_wmma_gemm<256, 256, 4, false, false, true><<<dim3(rowGroups, 4), B, 0, stream>>>(
        Abf, W3t, nullptr, nullptr, b3, alpha, 3, Ha, Hb, N);

    // ---- layer 4: SAGE linear. out = mean(h3)@W4 + h3@R4 + b4 ----
    zero(AGG, (long)N * 256);
    k_scatter<256><<<(int)cdivl((long)E * 64, B), B, 0, stream>>>(ei, Hb, nullptr, AGG, E);
    k_scale_cvt<256><<<(int)cdivl((long)N * 64, B), B, 0, stream>>>(AGG, cinv, Abf, N);
    k_cvt<<<(int)cdivl((long)N * 256, 4L * B), B, 0, stream>>>(Hb, Hbf, (long)N * 256);
    k_wmma_gemm<256, 112, 7, true, false, false><<<dim3(rowGroups, 1), B, 0, stream>>>(
        Abf, W4t, Hbf, R4t, b4, nullptr, 0, nullptr, (float*)d_out, N);
}
